// RNN_GAUSS_4294967296017
// MI455X (gfx1250) — matrix operand order repaired and checked by reading, not compiled, not run on hardware
//
#include <hip/hip_runtime.h>
#include <hip/hip_bf16.h>
#include <math.h>

typedef __bf16 bf16_t;
typedef __attribute__((ext_vector_type(8)))  __bf16 bfx8;
typedef __attribute__((ext_vector_type(16))) __bf16 bfx16;
typedef __attribute__((ext_vector_type(8)))  float  fx8;

#define LOG_SQRT_2PI 0.9189385332046727f

// ---------------- WMMA fragment helpers (contiguous-K layouts, wave32) ----------------

// A fragment: 16x32 bf16 tile from row-major [rows][ld].
// lane r=lane&15 -> row; lo-half lanes hold K 0..15, hi-half K 16..31 (contiguous).
__device__ __forceinline__ bfx16 frag_a(const bf16_t* __restrict__ base, int ld, int row0, int k0) {
  const int lane = threadIdx.x & 31;
  const int r  = lane & 15;
  const int hi = lane >> 4;
  const bf16_t* p = base + (size_t)(row0 + r) * ld + (k0 + hi * 16);
  bfx8 lo = *(const bfx8*)p;
  bfx8 hj = *(const bfx8*)(p + 8);
  bfx16 v;
#pragma unroll
  for (int i = 0; i < 8; ++i) { v[i] = lo[i]; v[i + 8] = hj[i]; }
  return v;
}

// B fragment: 32x16 bf16 tile, sourced from pre-transposed weights W^T stored [N][ld=K].
// lane c=lane&15 -> column; lo-half lanes hold K 0..15, hi-half K 16..31 (contiguous).
__device__ __forceinline__ bfx16 frag_b(const bf16_t* __restrict__ baseT, int ld, int col0, int k0) {
  const int lane = threadIdx.x & 31;
  const int c  = lane & 15;
  const int hi = lane >> 4;
  return *(const bfx16*)(baseT + (size_t)(col0 + c) * ld + (k0 + hi * 16));
}

__device__ __forceinline__ fx8 wmma_bf16(bfx16 a, bfx16 b, fx8 c) {
  return __builtin_amdgcn_wmma_f32_16x16x32_bf16(false, a, false, b, (short)0, c, false, false);
}

__device__ __forceinline__ fx8 zero8() {
  fx8 z;
#pragma unroll
  for (int i = 0; i < 8; ++i) z[i] = 0.0f;
  return z;
}

__device__ __forceinline__ float sigmoid_f(float x) { return 1.0f / (1.0f + expf(-x)); }
__device__ __forceinline__ float softplus_f(float x) { return (x > 20.0f) ? x : log1pf(expf(x)); }

// ---------------- Kernels ----------------

// out = relu(A[128xK] @ B[KxN] + bias), stored bf16. One wave per 16x16 tile.
__global__ __launch_bounds__(32) void gemm_relu_kernel(
    const bf16_t* __restrict__ A, const bf16_t* __restrict__ BT,
    const float* __restrict__ bias, bf16_t* __restrict__ out, int K, int N) {
  const int m0 = blockIdx.x * 16;
  const int n0 = blockIdx.y * 16;
  fx8 acc = zero8();
  for (int k0 = 0; k0 < K; k0 += 32) {
    bfx16 a = frag_a(A, K, m0, k0);
    bfx16 b = frag_b(BT, K, n0, k0);
    acc = wmma_bf16(a, b, acc);
  }
  const int lane = threadIdx.x & 31;
  const int n = n0 + (lane & 15);
  const int hi = lane >> 4;
  const float bv = bias[n];
#pragma unroll
  for (int i = 0; i < 8; ++i) {
    const int m = m0 + i + 8 * hi;
    float v = acc[i] + bv;
    v = v > 0.0f ? v : 0.0f;
    out[(size_t)m * N + n] = (bf16_t)v;
  }
}

// mean/std heads + Gaussian NLL partial sum into *out (atomic).
__global__ __launch_bounds__(32) void head_nll_kernel(
    const bf16_t* __restrict__ Dec, const bf16_t* __restrict__ WmT,
    const bf16_t* __restrict__ WsT, const float* __restrict__ bm,
    const float* __restrict__ bs, const float* __restrict__ Yt,  // [128][22] f32
    float* __restrict__ out) {
  const int bt = blockIdx.x & 7;
  const int nt = blockIdx.x >> 3;
  const int m0 = bt * 16, n0 = nt * 16;
  fx8 am = zero8(), as = zero8();
  for (int k0 = 0; k0 < 512; k0 += 32) {
    bfx16 a = frag_a(Dec, 512, m0, k0);
    am = wmma_bf16(a, frag_b(WmT, 512, n0, k0), am);
    as = wmma_bf16(a, frag_b(WsT, 512, n0, k0), as);
  }
  const int lane = threadIdx.x & 31;
  const int n = n0 + (lane & 15);
  const int hi = lane >> 4;
  float s = 0.0f;
  if (n < 22) {
    const float bmv = bm[n], bsv = bs[n];
#pragma unroll
    for (int i = 0; i < 8; ++i) {
      const int m = m0 + i + 8 * hi;
      const float mean = am[i] + bmv;
      const float sp = softplus_f(as[i] + bsv);
      const float d = (Yt[(size_t)m * 22 + n] - mean) / sp;
      s += 0.5f * d * d + logf(sp) + LOG_SQRT_2PI;
    }
  }
#pragma unroll
  for (int off = 16; off > 0; off >>= 1) s += __shfl_down(s, off, 32);
  if (lane == 0) atomicAdd(out, s);
}

// Fused GRU layer 0: gh = h@Wh (K=512), gx = y_t@Wx (K=32 padded), gate math, h update.
__global__ __launch_bounds__(32) void gru0_kernel(
    const bf16_t* __restrict__ Xt,    // [128][32] bf16 (padded y_t)
    const bf16_t* __restrict__ Hb,    // h0 bf16 in
    const float*  __restrict__ Hf,    // h0 f32 in
    const bf16_t* __restrict__ WxT,   // [1536][32]
    const bf16_t* __restrict__ WhT,   // [1536][512]
    const float* __restrict__ bx, const float* __restrict__ bh,
    bf16_t* __restrict__ Hob, float* __restrict__ Hof) {
  const int m0 = blockIdx.x * 16;
  const int c0 = blockIdx.y * 16;
  fx8 hr = zero8(), hz = zero8(), hn = zero8();
  for (int k0 = 0; k0 < 512; k0 += 32) {
    bfx16 a = frag_a(Hb, 512, m0, k0);
    hr = wmma_bf16(a, frag_b(WhT, 512, c0,        k0), hr);
    hz = wmma_bf16(a, frag_b(WhT, 512, c0 + 512,  k0), hz);
    hn = wmma_bf16(a, frag_b(WhT, 512, c0 + 1024, k0), hn);
  }
  fx8 xr = zero8(), xz = zero8(), xn = zero8();
  {
    bfx16 a = frag_a(Xt, 32, m0, 0);
    xr = wmma_bf16(a, frag_b(WxT, 32, c0,        0), xr);
    xz = wmma_bf16(a, frag_b(WxT, 32, c0 + 512,  0), xz);
    xn = wmma_bf16(a, frag_b(WxT, 32, c0 + 1024, 0), xn);
  }
  const int lane = threadIdx.x & 31;
  const int n = c0 + (lane & 15);
  const int hi = lane >> 4;
  const float bxr = bx[n], bxz = bx[n + 512], bxn = bx[n + 1024];
  const float bhr = bh[n], bhz = bh[n + 512], bhn = bh[n + 1024];
#pragma unroll
  for (int i = 0; i < 8; ++i) {
    const int m = m0 + i + 8 * hi;
    const float r  = sigmoid_f(xr[i] + bxr + hr[i] + bhr);
    const float z  = sigmoid_f(xz[i] + bxz + hz[i] + bhz);
    const float ng = tanhf(xn[i] + bxn + r * (hn[i] + bhn));
    const float ho = Hf[(size_t)m * 512 + n];
    const float hv = (1.0f - z) * ng + z * ho;
    Hof[(size_t)m * 512 + n] = hv;
    Hob[(size_t)m * 512 + n] = (bf16_t)hv;
  }
}

// Fused GRU layer 1: gx = h0_new@Wx (K=512), gh = h1@Wh (K=512), gate math, h update.
__global__ __launch_bounds__(32) void gru1_kernel(
    const bf16_t* __restrict__ H0n, const bf16_t* __restrict__ H1b,
    const float*  __restrict__ H1f,
    const bf16_t* __restrict__ WxT, const bf16_t* __restrict__ WhT,  // [1536][512]
    const float* __restrict__ bx, const float* __restrict__ bh,
    bf16_t* __restrict__ Hob, float* __restrict__ Hof) {
  const int m0 = blockIdx.x * 16;
  const int c0 = blockIdx.y * 16;
  fx8 xr = zero8(), xz = zero8(), xn = zero8();
  fx8 hr = zero8(), hz = zero8(), hn = zero8();
  for (int k0 = 0; k0 < 512; k0 += 32) {
    bfx16 a0 = frag_a(H0n, 512, m0, k0);
    bfx16 a1 = frag_a(H1b, 512, m0, k0);
    xr = wmma_bf16(a0, frag_b(WxT, 512, c0,        k0), xr);
    xz = wmma_bf16(a0, frag_b(WxT, 512, c0 + 512,  k0), xz);
    xn = wmma_bf16(a0, frag_b(WxT, 512, c0 + 1024, k0), xn);
    hr = wmma_bf16(a1, frag_b(WhT, 512, c0,        k0), hr);
    hz = wmma_bf16(a1, frag_b(WhT, 512, c0 + 512,  k0), hz);
    hn = wmma_bf16(a1, frag_b(WhT, 512, c0 + 1024, k0), hn);
  }
  const int lane = threadIdx.x & 31;
  const int n = c0 + (lane & 15);
  const int hi = lane >> 4;
  const float bxr = bx[n], bxz = bx[n + 512], bxn = bx[n + 1024];
  const float bhr = bh[n], bhz = bh[n + 512], bhn = bh[n + 1024];
#pragma unroll
  for (int i = 0; i < 8; ++i) {
    const int m = m0 + i + 8 * hi;
    const float r  = sigmoid_f(xr[i] + bxr + hr[i] + bhr);
    const float z  = sigmoid_f(xz[i] + bxz + hz[i] + bhz);
    const float ng = tanhf(xn[i] + bxn + r * (hn[i] + bhn));
    const float ho = H1f[(size_t)m * 512 + n];
    const float hv = (1.0f - z) * ng + z * ho;
    Hof[(size_t)m * 512 + n] = hv;
    Hob[(size_t)m * 512 + n] = (bf16_t)hv;
  }
}

// Transpose + convert weight src[K][N] f32 -> dst[Npad][Kpad] bf16 (zero-padded).
__global__ void conv_wt_kernel(const float* __restrict__ src, bf16_t* __restrict__ dst,
                               int K, int N, int Kpad, int Npad) {
  const int idx = blockIdx.x * 256 + threadIdx.x;
  if (idx >= Kpad * Npad) return;
  const int n = idx / Kpad;
  const int k = idx - n * Kpad;
  const float v = (n < N && k < K) ? src[(size_t)k * N + n] : 0.0f;
  dst[idx] = (bf16_t)v;
}

// states [T*B][22] f32 -> [T*B][32] bf16 (K-padded).
__global__ void conv_states_kernel(const float* __restrict__ src, bf16_t* __restrict__ dst,
                                   int total) {
  const int idx = blockIdx.x * 256 + threadIdx.x;
  if (idx >= total) return;
  const int row = idx >> 5;
  const int c = idx & 31;
  dst[idx] = (bf16_t)((c < 22) ? src[(size_t)row * 22 + c] : 0.0f);
}

__global__ void init_kernel(float* h0f, bf16_t* h0b, float* h1f, bf16_t* h1b, float* out) {
  const int idx = blockIdx.x * 256 + threadIdx.x;
  if (idx < 2 * 128 * 512) {
    h0f[idx] = 0.0f; h1f[idx] = 0.0f;
    h0b[idx] = (bf16_t)0.0f; h1b[idx] = (bf16_t)0.0f;
  }
  if (idx == 0) out[0] = 0.0f;
}

// ---------------- Host ----------------

static inline size_t align256(size_t x) { return (x + 255) & ~(size_t)255; }

extern "C" void kernel_launch(void* const* d_in, const int* in_sizes, int n_in,
                              void* d_out, int out_size, void* d_ws, size_t ws_size,
                              hipStream_t stream) {
  (void)in_sizes; (void)n_in; (void)out_size; (void)ws_size;
  const float* states = (const float*)d_in[0];
  const float* Wd1 = (const float*)d_in[1];  const float* bd1 = (const float*)d_in[2];
  const float* Wd2 = (const float*)d_in[3];  const float* bd2 = (const float*)d_in[4];
  const float* Wm  = (const float*)d_in[5];  const float* bm  = (const float*)d_in[6];
  const float* Wsw = (const float*)d_in[7];  const float* bsw = (const float*)d_in[8];
  const float* Wx0 = (const float*)d_in[9];  const float* Wh0 = (const float*)d_in[10];
  const float* bx0 = (const float*)d_in[11]; const float* bh0 = (const float*)d_in[12];
  const float* Wx1 = (const float*)d_in[13]; const float* Wh1 = (const float*)d_in[14];
  const float* bx1 = (const float*)d_in[15]; const float* bh1 = (const float*)d_in[16];
  float* out = (float*)d_out;

  char* w = (char*)d_ws;
  auto carve = [&](size_t bytes) -> char* { char* p = w; w += align256(bytes); return p; };
  bf16_t* Wd1T = (bf16_t*)carve(512 * 512 * 2);
  bf16_t* Wd2T = (bf16_t*)carve(512 * 512 * 2);
  bf16_t* WmT  = (bf16_t*)carve(32 * 512 * 2);
  bf16_t* WsT  = (bf16_t*)carve(32 * 512 * 2);
  bf16_t* Wx0T = (bf16_t*)carve(1536 * 32 * 2);
  bf16_t* Wh0T = (bf16_t*)carve(1536 * 512 * 2);
  bf16_t* Wx1T = (bf16_t*)carve(1536 * 512 * 2);
  bf16_t* Wh1T = (bf16_t*)carve(1536 * 512 * 2);
  bf16_t* Xbf  = (bf16_t*)carve((size_t)512 * 128 * 32 * 2);
  bf16_t* dec1 = (bf16_t*)carve(128 * 512 * 2);
  bf16_t* dec  = (bf16_t*)carve(128 * 512 * 2);
  float*  h0f  = (float*) carve(2 * 128 * 512 * 4);
  bf16_t* h0b  = (bf16_t*)carve(2 * 128 * 512 * 2);
  float*  h1f  = (float*) carve(2 * 128 * 512 * 4);
  bf16_t* h1b  = (bf16_t*)carve(2 * 128 * 512 * 2);

  auto convw = [&](const float* s, bf16_t* d, int K, int N, int Kp, int Np) {
    conv_wt_kernel<<<(Kp * Np + 255) / 256, 256, 0, stream>>>(s, d, K, N, Kp, Np);
  };
  convw(Wd1, Wd1T, 512, 512, 512, 512);
  convw(Wd2, Wd2T, 512, 512, 512, 512);
  convw(Wm,  WmT,  512, 22,  512, 32);
  convw(Wsw, WsT,  512, 22,  512, 32);
  convw(Wx0, Wx0T, 22, 1536, 32, 1536);
  convw(Wh0, Wh0T, 512, 1536, 512, 1536);
  convw(Wx1, Wx1T, 512, 1536, 512, 1536);
  convw(Wh1, Wh1T, 512, 1536, 512, 1536);
  {
    const int tot = 512 * 128 * 32;
    conv_states_kernel<<<(tot + 255) / 256, 256, 0, stream>>>(states, Xbf, tot);
  }
  init_kernel<<<(2 * 128 * 512 + 255) / 256, 256, 0, stream>>>(h0f, h0b, h1f, h1b, out);

  const dim3 blk(32, 1, 1);
  const dim3 gDec(8, 32, 1);   // 128x512 output tiles
  const dim3 gGru(8, 32, 1);   // 128x512 hidden tiles
  const size_t HP = 128 * 512; // parity stride (elements)

  for (int t = 0; t < 512; ++t) {
    const int pi = t & 1, po = pi ^ 1;
    // decoder from h1 (pre-update)
    gemm_relu_kernel<<<gDec, blk, 0, stream>>>(h1b + pi * HP, Wd1T, bd1, dec1, 512, 512);
    gemm_relu_kernel<<<gDec, blk, 0, stream>>>(dec1, Wd2T, bd2, dec, 512, 512);
    head_nll_kernel<<<16, blk, 0, stream>>>(dec, WmT, WsT, bm, bsw,
                                            states + (size_t)t * 128 * 22, out);
    // GRU layer 0
    gru0_kernel<<<gGru, blk, 0, stream>>>(Xbf + (size_t)t * 128 * 32,
                                          h0b + pi * HP, h0f + pi * HP,
                                          Wx0T, Wh0T, bx0, bh0,
                                          h0b + po * HP, h0f + po * HP);
    // GRU layer 1
    gru1_kernel<<<gGru, blk, 0, stream>>>(h0b + po * HP,
                                          h1b + pi * HP, h1f + pi * HP,
                                          Wx1T, Wh1T, bx1, bh1,
                                          h1b + po * HP, h1f + po * HP);
  }
}
